// PointerNetwork_88029649699224
// MI455X (gfx1250) — matrix operand order repaired and checked by reading, not compiled, not run on hardware
//
#include <hip/hip_runtime.h>
#include <hip/hip_bf16.h>

// ---------------------------------------------------------------------------
// Problem constants
// ---------------------------------------------------------------------------
constexpr int BB = 512;   // batch
constexpr int TT = 9;     // sequence length
constexpr int DD = 1024;  // input dim
constexpr int HH = 1024;  // hidden dim
constexpr int AATT = 1024;// attention dim
#define NEGINF (-100000000.0f)

typedef __attribute__((ext_vector_type(16))) __bf16 bf16x16;
typedef __attribute__((ext_vector_type(8)))  float  f32x8;

// ---------------------------------------------------------------------------
// Generic WMMA GEMM:  C(M,N) = A(M,K) @ W(N,K)^T [+ bias(N)]
//   A fp32 row-major, W bf16 row-major (N x K), C fp32 row-major.
//   M % 128 == 0, N % 64 == 0, K % 64 == 0 (true for all shapes here).
//   Block tile 128x64xBK64, 8 waves (4 along M x 2 along N), wave tile 32x32.
//   8 v_wmma per barrier period; all LDS fragment reads are ds_load_b128.
// ---------------------------------------------------------------------------
#define BK  64
#define AST 72   // LDS row stride (bf16 elems); 144 B per row keeps 16B align
#define BST 72

__global__ __launch_bounds__(256)
void gemm_bf16_wmma(float* __restrict__ C, const float* __restrict__ A,
                    const __bf16* __restrict__ W, const float* __restrict__ bias,
                    int M, int N, int K) {
  __shared__ alignas(16) __bf16 As[128 * AST];
  __shared__ alignas(16) __bf16 Bs[64 * BST];

  const int tid   = threadIdx.x;
  const int wid   = tid >> 5;
  const int lane  = tid & 31;
  const int nBlk  = N >> 6;
  const int m0    = (blockIdx.x / nBlk) << 7;
  const int n0    = (blockIdx.x % nBlk) << 6;
  const int waveM = wid >> 1;   // 0..3
  const int waveN = wid & 1;    // 0..1

  // staging assignments (BK=64 slab)
  const int arow = tid >> 1;           // 0..127
  const int acol = (tid & 1) << 5;     // 0 or 32  (32 floats per thread)
  const int brow = tid >> 2;           // 0..63
  const int bcol = (tid & 3) << 4;     // 0,16,32,48 (16 bf16 per thread)

  const float*  aSrcBase = A + (size_t)(m0 + arow) * K + acol;
  const __bf16* bSrcBase = W + (size_t)(n0 + brow) * K + bcol;

  // fragment addressing (documented gfx1250 WMMA VGPR layouts, §7.12.2)
  const int am  = lane & 15;
  const int akb = (lane >> 4) << 3;   // A: 0 or 8
  const int bnn = lane & 15;
  const int bkb = (lane >> 4) << 3;   // B: 0 or 8

  f32x8 acc[2][2] = {};

  for (int k0 = 0; k0 < K; k0 += BK) {
    // ---- stage A slab (fp32 -> bf16) into LDS --------------------------
    {
      const float4* src = (const float4*)(aSrcBase + k0);
      __bf16* dst = &As[arow * AST + acol];
#pragma unroll
      for (int q = 0; q < 8; ++q) {
        float4 f = src[q];
        dst[q * 4 + 0] = (__bf16)f.x;
        dst[q * 4 + 1] = (__bf16)f.y;
        dst[q * 4 + 2] = (__bf16)f.z;
        dst[q * 4 + 3] = (__bf16)f.w;
      }
      // speculative prefetch of next slab (safe past end: silently dropped)
      __builtin_prefetch((const char*)src + BK * 4, 0, 1);
    }
    // ---- stage W slab (bf16) into LDS ----------------------------------
    {
      const uint4* src = (const uint4*)(bSrcBase + k0);
      uint4* dst = (uint4*)&Bs[brow * BST + bcol];
      dst[0] = src[0];
      dst[1] = src[1];
      __builtin_prefetch((const char*)src + BK * 2, 0, 1);
    }
    __syncthreads();

#pragma unroll
    for (int kk = 0; kk < BK; kk += 32) {
      // A (16x32 bf16): lanes 0-15 row M=lane hold K {0..7,16..23};
      //                 lanes 16-31 row M=lane-16 hold K {8..15,24..31}
      bf16x16 afrag[2];
#pragma unroll
      for (int i = 0; i < 2; ++i) {
        const __bf16* base = &As[(waveM * 32 + i * 16 + am) * AST + kk];
        union { uint4 u[2]; bf16x16 v; } u;
        u.u[0] = *(const uint4*)(base + akb);
        u.u[1] = *(const uint4*)(base + akb + 16);
        afrag[i] = u.v;
      }
      // B (32x16 bf16): lanes 0-15 col N=lane hold K {0..7,16..23};
      //                 lanes 16-31 col N=lane-16 hold K {8..15,24..31}
      bf16x16 bfrag[2];
#pragma unroll
      for (int j = 0; j < 2; ++j) {
        const __bf16* base = &Bs[(waveN * 32 + j * 16 + bnn) * BST + kk];
        union { uint4 u[2]; bf16x16 v; } u;
        u.u[0] = *(const uint4*)(base + bkb);
        u.u[1] = *(const uint4*)(base + bkb + 16);
        bfrag[j] = u.v;
      }
#pragma unroll
      for (int i = 0; i < 2; ++i)
#pragma unroll
        for (int j = 0; j < 2; ++j)
          acc[i][j] = __builtin_amdgcn_wmma_f32_16x16x32_bf16(
              false, afrag[i], false, bfrag[j], (short)0, acc[i][j], false, false);
    }
    __syncthreads();
  }

  // ---- epilogue: C/D layout: lanes 0-15 M=0..7, lanes 16-31 M=8..15 ------
  const int cn = lane & 15;
  const int cm = (lane >> 4) << 3;
#pragma unroll
  for (int i = 0; i < 2; ++i) {
#pragma unroll
    for (int j = 0; j < 2; ++j) {
      const int col = n0 + waveN * 32 + j * 16 + cn;
      const float bv = bias ? bias[col] : 0.0f;
      const int rowBase = m0 + waveM * 32 + i * 16 + cm;
#pragma unroll
      for (int r = 0; r < 8; ++r)
        C[(size_t)(rowBase + r) * N + col] = acc[i][j][r] + bv;
    }
  }
}

// ---------------------------------------------------------------------------
// Elementwise / glue kernels
// ---------------------------------------------------------------------------
// n must be a multiple of 4 (true for every weight matrix here)
__global__ void cvt_f32_bf16(__bf16* __restrict__ dst, const float* __restrict__ src, long n4) {
  long i = (long)blockIdx.x * blockDim.x + threadIdx.x;
  if (i < n4) {
    float4 f = ((const float4*)src)[i];
    __bf16* d = dst + i * 4;
    d[0] = (__bf16)f.x; d[1] = (__bf16)f.y; d[2] = (__bf16)f.z; d[3] = (__bf16)f.w;
  }
}

__global__ void zero_f32(float* __restrict__ p, long n) {
  long i = (long)blockIdx.x * blockDim.x + threadIdx.x;
  if (i < n) p[i] = 0.0f;
}

// GRU gate: h := (1-z)*n + z*h, with gi already = x@Wih^T+bih, gh = h@Whh^T+bhh.
// gi row for batch b is gi + b*giStrideB (handles X's (b,t) interleave).
// Optionally mirrors the new h into y (encoder concat output), y + b*yStride + j.
__global__ void gru_gate(const float* __restrict__ gi, long giStrideB,
                         const float* __restrict__ gh, float* __restrict__ h,
                         float* __restrict__ y, long yStrideB, int B, int Hh) {
  long idx = (long)blockIdx.x * blockDim.x + threadIdx.x;
  if (idx >= (long)B * Hh) return;
  int b = (int)(idx / Hh), j = (int)(idx % Hh);
  const float* gib = gi + (long)b * giStrideB;
  const float* ghb = gh + (long)b * 3 * Hh;
  float ir = gib[j], iz = gib[Hh + j], in_ = gib[2 * Hh + j];
  float hr = ghb[j], hz = ghb[Hh + j], hn = ghb[2 * Hh + j];
  float r = 1.0f / (1.0f + __expf(-(ir + hr)));
  float z = 1.0f / (1.0f + __expf(-(iz + hz)));
  float n = tanhf(in_ + r * hn);
  float hp = h[(long)b * Hh + j];
  float hnew = (1.0f - z) * n + z * hp;
  h[(long)b * Hh + j] = hnew;
  if (y) y[(long)b * yStrideB + j] = hnew;
}

// finals: hdst(b, 2H) = concat(y[T-1][b][0:H], y[0][b][H:2H]); y is (T*B, 2H)
__global__ void finals_concat(float* __restrict__ hdst, const float* __restrict__ y,
                              int B, int Hh) {
  long idx = (long)blockIdx.x * blockDim.x + threadIdx.x;
  if (idx >= (long)B * 2 * Hh) return;
  int b = (int)(idx / (2 * Hh)), j = (int)(idx % (2 * Hh));
  float val = (j < Hh) ? y[((size_t)(TT - 1) * B + b) * 2 * Hh + j]
                       : y[((size_t)b) * 2 * Hh + j];
  hdst[idx] = val;
}

__global__ void bcast_start(float* __restrict__ dst, const float* __restrict__ src, int B, int Dd) {
  long idx = (long)blockIdx.x * blockDim.x + threadIdx.x;
  if (idx >= (long)B * Dd) return;
  dst[idx] = src[idx % Dd];
}

// Attention scores: one wave32 per (b,t); ui = tanh(w2d[b]+W1e[t*B+b]) . v
// Masked score written straight into d_out[b][step][t].
__global__ __launch_bounds__(256)
void attn_scores(float* __restrict__ out, const float* __restrict__ w2d,
                 const float* __restrict__ W1e, const float* __restrict__ v,
                 const unsigned* __restrict__ mask, int step) {
  int wid = threadIdx.x >> 5, lane = threadIdx.x & 31;
  int g = blockIdx.x * 8 + wid;          // g = t*B + b
  int t = g >> 9;                        // /512
  int b = g & 511;
  const float* we = W1e + (size_t)g * AATT;
  const float* wd = w2d + (size_t)b * AATT;
  float s = 0.0f;
  for (int a = lane; a < AATT; a += 32)
    s += tanhf(wd[a] + we[a]) * v[a];
#pragma unroll
  for (int off = 16; off; off >>= 1) s += __shfl_xor(s, off, 32);
  if (lane == 0) {
    if ((mask[b] >> t) & 1u) s = NEGINF;
    out[((size_t)b * TT + step) * TT + t] = s;
  }
}

// Greedy pointer: argmax over masked logits row, update mask, gather next input.
__global__ void argmax_gather(float* __restrict__ dec_in, unsigned* __restrict__ mask,
                              const float* __restrict__ logits, const float* __restrict__ X,
                              int step) {
  int b = blockIdx.x;
  __shared__ int sBest;
  if (threadIdx.x == 0) {
    const float* row = logits + ((size_t)b * TT + step) * TT;
    int best = 0; float bv = row[0];
    for (int t = 1; t < TT; ++t) { float vv = row[t]; if (vv > bv) { bv = vv; best = t; } }
    mask[b] |= (1u << best);
    sBest = best;
  }
  __syncthreads();
  const float* src = X + ((size_t)b * TT + sBest) * DD;
  float* dst = dec_in + (size_t)b * DD;
  for (int d = threadIdx.x; d < DD; d += blockDim.x) dst[d] = src[d];
}

// ---------------------------------------------------------------------------
// Host orchestration
// ---------------------------------------------------------------------------
static inline size_t alignUp(size_t x) { return (x + 255) & ~(size_t)255; }

static void launch_gemm(float* C, const float* A, const __bf16* W, const float* bias,
                        int M, int N, int K, hipStream_t s) {
  dim3 grid((M / 128) * (N / 64));
  gemm_bf16_wmma<<<grid, 256, 0, s>>>(C, A, W, bias, M, N, K);
}
static void launch_cvt(__bf16* dst, const float* src, long n, hipStream_t s) {
  long n4 = n / 4;
  cvt_f32_bf16<<<(unsigned)((n4 + 255) / 256), 256, 0, s>>>(dst, src, n4);
}
static void launch_zero(float* p, long n, hipStream_t s) {
  zero_f32<<<(unsigned)((n + 255) / 256), 256, 0, s>>>(p, n);
}
static void launch_gate(const float* gi, long giStrideB, const float* gh, float* h,
                        float* y, long yStride, int B, int Hh, hipStream_t s) {
  long n = (long)B * Hh;
  gru_gate<<<(unsigned)((n + 255) / 256), 256, 0, s>>>(gi, giStrideB, gh, h, y, yStride, B, Hh);
}

extern "C" void kernel_launch(void* const* d_in, const int* in_sizes, int n_in,
                              void* d_out, int out_size, void* d_ws, size_t ws_size,
                              hipStream_t stream) {
  (void)in_sizes; (void)n_in; (void)out_size; (void)ws_size;

  const float* X = (const float*)d_in[0];
  // encoder weights: idx = 1 + 4*scan  (scan: 0=l0f,1=l0b,2=l1f,3=l1b)
  const float* eWih[4] = { (const float*)d_in[1],  (const float*)d_in[5],
                           (const float*)d_in[9],  (const float*)d_in[13] };
  const float* eWhh[4] = { (const float*)d_in[2],  (const float*)d_in[6],
                           (const float*)d_in[10], (const float*)d_in[14] };
  const float* eBih[4] = { (const float*)d_in[3],  (const float*)d_in[7],
                           (const float*)d_in[11], (const float*)d_in[15] };
  const float* eBhh[4] = { (const float*)d_in[4],  (const float*)d_in[8],
                           (const float*)d_in[12], (const float*)d_in[16] };
  const float* dWih[2] = { (const float*)d_in[17], (const float*)d_in[21] };
  const float* dWhh[2] = { (const float*)d_in[18], (const float*)d_in[22] };
  const float* dBih[2] = { (const float*)d_in[19], (const float*)d_in[23] };
  const float* dBhh[2] = { (const float*)d_in[20], (const float*)d_in[24] };
  const float* W1 = (const float*)d_in[25];
  const float* W2 = (const float*)d_in[26];
  const float* v  = (const float*)d_in[27];
  const float* start = (const float*)d_in[28];
  float* out = (float*)d_out;

  // ---- workspace carve-out (deterministic) ------------------------------
  char* p = (char*)d_ws;
  auto carve = [&](size_t bytes) { void* r = (void*)p; p += alignUp(bytes); return r; };

  const long eWihN[4] = { 3072L * 1024, 3072L * 1024, 3072L * 2048, 3072L * 2048 };
  const long eWhhN   = 3072L * 1024;
  const long dWihN[2] = { 6144L * 1024, 6144L * 2048 };
  const long dWhhN[2] = { 6144L * 2048, 6144L * 2048 };

  __bf16* eWihB[4]; __bf16* eWhhB[4]; __bf16* dWihB[2]; __bf16* dWhhB[2];
  for (int i = 0; i < 4; ++i) { eWihB[i] = (__bf16*)carve(eWihN[i] * 2); eWhhB[i] = (__bf16*)carve(eWhhN * 2); }
  for (int i = 0; i < 2; ++i) { dWihB[i] = (__bf16*)carve(dWihN[i] * 2); dWhhB[i] = (__bf16*)carve(dWhhN[i] * 2); }
  __bf16* W1b = (__bf16*)carve(1024L * 2048 * 2);
  __bf16* W2b = (__bf16*)carve(1024L * 2048 * 2);

  float* giBig  = (float*)carve((size_t)TT * BB * 3072 * 4);  // reused across 4 scans
  float* y0     = (float*)carve((size_t)TT * BB * 2048 * 4);
  float* y1     = (float*)carve((size_t)TT * BB * 2048 * 4);
  float* ghBuf  = (float*)carve((size_t)BB * 6144 * 4);
  float* giDec  = (float*)carve((size_t)BB * 6144 * 4);
  float* hbuf   = (float*)carve((size_t)BB * HH * 4);
  float* W1e    = (float*)carve((size_t)TT * BB * AATT * 4);
  float* h0d    = (float*)carve((size_t)BB * 2 * HH * 4);
  float* h1d    = (float*)carve((size_t)BB * 2 * HH * 4);
  float* dec_in = (float*)carve((size_t)BB * DD * 4);
  float* w2d    = (float*)carve((size_t)BB * AATT * 4);
  unsigned* mask = (unsigned*)carve(BB * 4);

  // ---- convert weights to bf16 (≈160 MB total; fits global L2) ----------
  for (int i = 0; i < 4; ++i) {
    launch_cvt(eWihB[i], eWih[i], eWihN[i], stream);
    launch_cvt(eWhhB[i], eWhh[i], eWhhN, stream);
  }
  for (int i = 0; i < 2; ++i) {
    launch_cvt(dWihB[i], dWih[i], dWihN[i], stream);
    launch_cvt(dWhhB[i], dWhh[i], dWhhN[i], stream);
  }
  launch_cvt(W1b, W1, 1024L * 2048, stream);
  launch_cvt(W2b, W2, 1024L * 2048, stream);

  // ---- encoder ----------------------------------------------------------
  // scan 0: layer0 fwd, scan 1: layer0 bwd, scan 2: layer1 fwd, scan 3: layer1 bwd
  for (int scan = 0; scan < 4; ++scan) {
    const int layer = scan >> 1;
    const bool bwd  = scan & 1;
    float* yOut = layer ? y1 : y0;
    if (layer == 0) {
      // gi = X @ Wih^T + bih  over all (b,t); X rows are b*T + t
      launch_gemm(giBig, X, eWihB[scan], eBih[scan], BB * TT, 3072, DD, stream);
    } else {
      // gi = y0 @ Wih^T + bih; y0 rows are t*B + b
      launch_gemm(giBig, y0, eWihB[scan], eBih[scan], BB * TT, 3072, 2 * HH, stream);
    }
    launch_zero(hbuf, (long)BB * HH, stream);
    for (int k = 0; k < TT; ++k) {
      const int t = bwd ? (TT - 1 - k) : k;
      launch_gemm(ghBuf, hbuf, eWhhB[scan], eBhh[scan], BB, 3072, HH, stream);
      const float* giT; long giStride;
      if (layer == 0) { giT = giBig + (size_t)t * 3072;            giStride = (long)TT * 3072; }
      else            { giT = giBig + (size_t)t * BB * 3072;       giStride = 3072; }
      float* yT = yOut + (size_t)t * BB * 2048 + (bwd ? HH : 0);
      launch_gate(giT, giStride, ghBuf, hbuf, yT, 2048, BB, HH, stream);
    }
  }

  // W1_e = enc_out @ W1^T (rows ordered t*B + b), no bias
  launch_gemm(W1e, y1, W1b, nullptr, BB * TT, AATT, 2 * HH, stream);

  // decoder initial hiddens = per-layer final states (fwd tail, bwd head)
  {
    long n = (long)BB * 2 * HH;
    finals_concat<<<(unsigned)((n + 255) / 256), 256, 0, stream>>>(h0d, y0, BB, HH);
    finals_concat<<<(unsigned)((n + 255) / 256), 256, 0, stream>>>(h1d, y1, BB, HH);
    long m = (long)BB * DD;
    bcast_start<<<(unsigned)((m + 255) / 256), 256, 0, stream>>>(dec_in, start, BB, DD);
    launch_zero((float*)mask, BB, stream);  // zero bits == zero float
  }

  // ---- decoder: 9 greedy pointer steps ----------------------------------
  for (int step = 0; step < TT; ++step) {
    // layer 0 cell (hidden 2H)
    launch_gemm(giDec, dec_in, dWihB[0], dBih[0], BB, 6144, DD, stream);
    launch_gemm(ghBuf, h0d,   dWhhB[0], dBhh[0], BB, 6144, 2 * HH, stream);
    launch_gate(giDec, 6144, ghBuf, h0d, nullptr, 0, BB, 2 * HH, stream);
    // layer 1 cell (input = new h0d)
    launch_gemm(giDec, h0d, dWihB[1], dBih[1], BB, 6144, 2 * HH, stream);
    launch_gemm(ghBuf, h1d, dWhhB[1], dBhh[1], BB, 6144, 2 * HH, stream);
    launch_gate(giDec, 6144, ghBuf, h1d, nullptr, 0, BB, 2 * HH, stream);
    // attention query
    launch_gemm(w2d, h1d, W2b, nullptr, BB, AATT, 2 * HH, stream);
    // masked scores -> d_out[b][step][t]
    attn_scores<<<(BB * TT) / 8, 256, 0, stream>>>(out, w2d, W1e, v, mask, step);
    // greedy argmax, mask update, gather next decoder input
    argmax_gather<<<BB, 256, 0, stream>>>(dec_in, mask, out, X, step);
  }
}
